// MultiHeadAttention_23880018165773
// MI455X (gfx1250) — matrix operand order repaired and checked by reading, not compiled, not run on hardware
//
#include <hip/hip_runtime.h>

#define HIDDEN   1024
#define HEADS    16
#define HEAD_DIM 64
#define BATCH    4
#define SEQ      2048
#define MTOT     (BATCH * SEQ)

typedef __bf16 bf16_t;
typedef __attribute__((ext_vector_type(16))) __bf16 v16bf;
typedef __attribute__((ext_vector_type(8)))  float v8f;
typedef __attribute__((ext_vector_type(4)))  int   v4i;

#if defined(__has_builtin)
#if __has_builtin(__builtin_amdgcn_global_load_async_to_lds_b128)
#define HAVE_ASYNC_LDS 1
#endif
#if __has_builtin(__builtin_amdgcn_s_wait_asynccnt)
#define HAVE_WAIT_ASYNC_BUILTIN 1
#endif
#endif

#ifdef HAVE_ASYNC_LDS
#define AS1 __attribute__((address_space(1)))
#define AS3 __attribute__((address_space(3)))
#endif

// ---- WMMA fragment index maps (CDNA5 ISA §7.12.2, wave32) ----
// 16-bit A matrix 16x32: lane L holds row M=L%16; element i of the v16bf
// maps to K = i + 16*(L>=16) (same contiguous order as the B fragment),
// so LDS reads merge into ds_load_b128 pairs.
static __device__ __forceinline__ int a_kmap(int i, int half) {
  return i + (half ? 16 : 0);
}

// B fragment (32x16, 16-bit): lane L holds col N=L%16, K = i + 16*(L>=16).
// Caller arranges LDS so the 16 K-values are contiguous at `base`.
static __device__ __forceinline__ v16bf loadB_contig(const bf16_t* base,
                                                     int half) {
  v16bf b;
#pragma unroll
  for (int i = 0; i < 16; ++i) b[i] = base[(half ? 16 : 0) + i];
  return b;
}

// Row-wise reductions over the 16 lanes of a wave half (C/D layout keeps a
// row's 16 columns on lanes [16*half .. 16*half+15], same VGPR).
static __device__ __forceinline__ float row_max16(float v) {
  v = fmaxf(v, __shfl_xor(v, 1, 32));
  v = fmaxf(v, __shfl_xor(v, 2, 32));
  v = fmaxf(v, __shfl_xor(v, 4, 32));
  v = fmaxf(v, __shfl_xor(v, 8, 32));
  return v;
}
static __device__ __forceinline__ float row_sum16(float v) {
  v += __shfl_xor(v, 1, 32);
  v += __shfl_xor(v, 2, 32);
  v += __shfl_xor(v, 4, 32);
  v += __shfl_xor(v, 8, 32);
  return v;
}

static __device__ __forceinline__ v8f wmma_bf16(v16bf a, v16bf b, v8f c) {
  return __builtin_amdgcn_wmma_f32_16x16x32_bf16(
      /*neg_a=*/false, a, /*neg_b=*/false, b,
      /*c_mod=*/(short)0, c, /*reuse_a=*/false, /*reuse_b=*/false);
}

// =====================  QKV projection: f32 in, bf16 out  ==================
// C[M,N] = X[M,K] @ W[K,N] + bias; out scattered to [B, H, S, D] bf16.
// LDS: Xt row-major [64][32] (A reads contiguous); Wt TRANSPOSED [col][k]
// so B reads are contiguous.
__global__ __launch_bounds__(128) void mha_proj_qkv(
    const float* __restrict__ X, const float* __restrict__ W,
    const float* __restrict__ bias, bf16_t* __restrict__ out) {
  __shared__ bf16_t Xt[64 * 32];  // [row][k]
  __shared__ bf16_t Wt[64 * 32];  // [col][k]  (transposed)
  const int tid  = threadIdx.x;
  const int lane = tid & 31;
  const int wave = tid >> 5;
  const int half = lane >> 4;
  const int ln   = lane & 15;
  const int m_base = blockIdx.x * 64;
  const int n_base = blockIdx.y * 64;

  v8f acc[4] = {};
  for (int k0 = 0; k0 < HIDDEN; k0 += 32) {
    __syncthreads();
#pragma unroll
    for (int i = 0; i < 16; ++i) {           // stage X tile (f32 -> bf16)
      int idx = tid * 16 + i;                // contiguous per-thread run
      int r = idx >> 5, c = idx & 31;
      Xt[idx] = (bf16_t)X[(m_base + r) * HIDDEN + k0 + c];
    }
#pragma unroll
    for (int i = 0; i < 16; ++i) {           // stage W tile transposed
      int idx = tid * 16 + i;                // row-contiguous global read
      int r = idx >> 6, c = idx & 63;        // r = k, c = col
      Wt[c * 32 + r] = (bf16_t)W[(k0 + r) * HIDDEN + n_base + c];
    }
    __syncthreads();
    v16bf a;
#pragma unroll
    for (int i = 0; i < 16; ++i)
      a[i] = Xt[(wave * 16 + ln) * 32 + a_kmap(i, half)];
#pragma unroll
    for (int n = 0; n < 4; ++n) {
      v16bf b = loadB_contig(&Wt[(n * 16 + ln) * 32], half);
      acc[n] = wmma_bf16(a, b, acc[n]);
    }
  }
  // Write with bias, scattered to [B,H,S,D]
#pragma unroll
  for (int n = 0; n < 4; ++n) {
    int col = n_base + n * 16 + ln;
    int h = col >> 6, d = col & 63;
    float bv = bias[col];
#pragma unroll
    for (int r = 0; r < 8; ++r) {
      int m  = m_base + wave * 16 + r + 8 * half;
      int bb = m / SEQ, s = m % SEQ;
      out[(((size_t)bb * HEADS + h) * SEQ + s) * HEAD_DIM + d] =
          (bf16_t)(acc[n][r] + bv);
    }
  }
}

// =====================  Flash attention over bf16 Q/K/V  ===================
__global__ __launch_bounds__(128) void mha_attn(
    const bf16_t* __restrict__ Qw, const bf16_t* __restrict__ Kw,
    const bf16_t* __restrict__ Vw, bf16_t* __restrict__ O) {
  __shared__ bf16_t Kt[32 * 64];     // [kv][d]   (B-frag contiguous in d)
  __shared__ bf16_t Vt[64 * 32];     // [d][kv]   (transposed: contiguous kv)
  __shared__ bf16_t Pt[4][16 * 32];  // per-wave P restripe buffer
  const int tid  = threadIdx.x;
  const int lane = tid & 31;
  const int wave = tid >> 5;
  const int half = lane >> 4;
  const int ln   = lane & 15;
  const int bh   = blockIdx.y;                // b*HEADS + h
  const int q0   = blockIdx.x * 64 + wave * 16;

  const bf16_t* Qh = Qw + (size_t)bh * SEQ * HEAD_DIM;
  const bf16_t* Kh = Kw + (size_t)bh * SEQ * HEAD_DIM;
  const bf16_t* Vh = Vw + (size_t)bh * SEQ * HEAD_DIM;

  // Q fragments live in registers for the whole kv sweep (2 chunks of K=32)
  v16bf aq[2];
#pragma unroll
  for (int c = 0; c < 2; ++c)
#pragma unroll
    for (int i = 0; i < 16; ++i)
      aq[c][i] = Qh[(q0 + ln) * HEAD_DIM + c * 32 + a_kmap(i, half)];

  float m_r[8], l_r[8];
#pragma unroll
  for (int r = 0; r < 8; ++r) { m_r[r] = -1e30f; l_r[r] = 0.0f; }
  v8f acc[4] = {};

  const float scale = 0.125f;  // 1/sqrt(HEAD_DIM)

  for (int kv = 0; kv < SEQ; kv += 32) {
    __syncthreads();
    {  // stage K tile (straight copy, 256 x 16B) -- async if available
      const uint4* gk = (const uint4*)(Kh + (size_t)kv * HEAD_DIM);
      uint4* sk = (uint4*)Kt;
#ifdef HAVE_ASYNC_LDS
      __builtin_amdgcn_global_load_async_to_lds_b128(
          (AS1 v4i*)(gk + tid * 2 + 0), (AS3 v4i*)(sk + tid * 2 + 0), 0, 0);
      __builtin_amdgcn_global_load_async_to_lds_b128(
          (AS1 v4i*)(gk + tid * 2 + 1), (AS3 v4i*)(sk + tid * 2 + 1), 0, 0);
#else
      sk[tid * 2 + 0] = gk[tid * 2 + 0];
      sk[tid * 2 + 1] = gk[tid * 2 + 1];
#endif
    }
    {  // stage V tile transposed: read 16 contiguous bf16, scatter by column
      int kvr = tid >> 2;            // 0..31
      int d0  = (tid & 3) * 16;
      const bf16_t* src = Vh + (size_t)(kv + kvr) * HEAD_DIM + d0;
#pragma unroll
      for (int i = 0; i < 16; ++i) Vt[(d0 + i) * 32 + kvr] = src[i];
    }
    if (kv + 32 < SEQ) {  // global_prefetch_b8 of next tiles
      __builtin_prefetch(Kh + (size_t)(kv + 32) * HEAD_DIM + tid * 16, 0, 1);
      __builtin_prefetch(Vh + (size_t)(kv + 32) * HEAD_DIM + tid * 16, 0, 1);
    }
#ifdef HAVE_ASYNC_LDS
#ifdef HAVE_WAIT_ASYNC_BUILTIN
    __builtin_amdgcn_s_wait_asynccnt(0);
#else
    asm volatile("s_wait_asynccnt 0" ::: "memory");
#endif
#endif
    __syncthreads();

    // S = Q @ K^T : 2 column chunks x 2 K-dim chunks
    v8f s[2] = {};
#pragma unroll
    for (int nc = 0; nc < 2; ++nc)
#pragma unroll
      for (int dc = 0; dc < 2; ++dc) {
        // B element i: K=dc*32 + (half?16:0)+i, col = nc*16+ln -> Kt[col][K]
        v16bf b = loadB_contig(&Kt[(nc * 16 + ln) * 64 + dc * 32], half);
        s[nc] = wmma_bf16(aq[dc], b, s[nc]);
      }
#pragma unroll
    for (int nc = 0; nc < 2; ++nc)
#pragma unroll
      for (int r = 0; r < 8; ++r) s[nc][r] *= scale;

    // online softmax per row r (row = r + 8*half)
#pragma unroll
    for (int r = 0; r < 8; ++r) {
      float rm    = row_max16(fmaxf(s[0][r], s[1][r]));
      float mnew  = fmaxf(m_r[r], rm);
      float alpha = __expf(m_r[r] - mnew);
      float p0    = __expf(s[0][r] - mnew);
      float p1    = __expf(s[1][r] - mnew);
      s[0][r] = p0;
      s[1][r] = p1;
      l_r[r] = l_r[r] * alpha + row_sum16(p0 + p1);
      m_r[r] = mnew;
#pragma unroll
      for (int n = 0; n < 4; ++n) acc[n][r] *= alpha;
    }

    // restripe P (C-layout -> A-layout) through per-wave LDS
    bf16_t* pt = Pt[wave];
#pragma unroll
    for (int nc = 0; nc < 2; ++nc)
#pragma unroll
      for (int r = 0; r < 8; ++r)
        pt[(r + 8 * half) * 32 + nc * 16 + ln] = (bf16_t)s[nc][r];
    asm volatile("s_wait_dscnt 0" ::: "memory");
    v16bf ap;
#pragma unroll
    for (int i = 0; i < 16; ++i) ap[i] = pt[ln * 32 + a_kmap(i, half)];

    // acc += P @ V   (4 d-chunks of 16 cols; Vt[d][kv] -> contiguous kv)
#pragma unroll
    for (int n = 0; n < 4; ++n) {
      v16bf bv = loadB_contig(&Vt[(n * 16 + ln) * 32], half);
      acc[n] = wmma_bf16(ap, bv, acc[n]);
    }
  }

  // normalize and write attn output as [B, S, HIDDEN] bf16 (h*64+d columns)
  const int h  = bh % HEADS;
  const int bb = bh / HEADS;
#pragma unroll
  for (int n = 0; n < 4; ++n)
#pragma unroll
    for (int r = 0; r < 8; ++r) {
      int row = q0 + r + 8 * half;
      float o = acc[n][r] / l_r[r];
      O[((size_t)(bb * SEQ + row)) * HIDDEN + h * HEAD_DIM + n * 16 + ln] =
          (bf16_t)o;
    }
}

// ==============  Output projection: bf16 in, f32 out (+bias)  ==============
__global__ __launch_bounds__(128) void mha_proj_out(
    const bf16_t* __restrict__ X, const float* __restrict__ W,
    const float* __restrict__ bias, float* __restrict__ out) {
  __shared__ bf16_t Xt[64 * 32];  // [row][k]
  __shared__ bf16_t Wt[64 * 32];  // [col][k]  (transposed)
  const int tid  = threadIdx.x;
  const int lane = tid & 31;
  const int wave = tid >> 5;
  const int half = lane >> 4;
  const int ln   = lane & 15;
  const int m_base = blockIdx.x * 64;
  const int n_base = blockIdx.y * 64;

  v8f acc[4] = {};
  for (int k0 = 0; k0 < HIDDEN; k0 += 32) {
    __syncthreads();
#pragma unroll
    for (int i = 0; i < 16; ++i) {
      int idx = tid * 16 + i;
      int r = idx >> 5, c = idx & 31;
      Xt[idx] = X[(m_base + r) * HIDDEN + k0 + c];
    }
#pragma unroll
    for (int i = 0; i < 16; ++i) {
      int idx = tid * 16 + i;
      int r = idx >> 6, c = idx & 63;
      Wt[c * 32 + r] = (bf16_t)W[(k0 + r) * HIDDEN + n_base + c];
    }
    __syncthreads();
    v16bf a;
#pragma unroll
    for (int i = 0; i < 16; ++i)
      a[i] = Xt[(wave * 16 + ln) * 32 + a_kmap(i, half)];
#pragma unroll
    for (int n = 0; n < 4; ++n) {
      v16bf b = loadB_contig(&Wt[(n * 16 + ln) * 32], half);
      acc[n] = wmma_bf16(a, b, acc[n]);
    }
  }
#pragma unroll
  for (int n = 0; n < 4; ++n) {
    int col = n_base + n * 16 + ln;
    float bv = bias[col];
#pragma unroll
    for (int r = 0; r < 8; ++r) {
      int m = m_base + wave * 16 + r + 8 * half;
      out[(size_t)m * HIDDEN + col] = acc[n][r] + bv;
    }
  }
}

extern "C" void kernel_launch(void* const* d_in, const int* in_sizes, int n_in,
                              void* d_out, int out_size, void* d_ws,
                              size_t ws_size, hipStream_t stream) {
  (void)in_sizes; (void)n_in; (void)out_size; (void)ws_size;
  const float* queries = (const float*)d_in[0];
  const float* keys    = (const float*)d_in[1];
  const float* values  = (const float*)d_in[2];
  const float* Wq = (const float*)d_in[3];
  const float* bq = (const float*)d_in[4];
  const float* Wk = (const float*)d_in[5];
  const float* bk = (const float*)d_in[6];
  const float* Wv = (const float*)d_in[7];
  const float* bv = (const float*)d_in[8];
  const float* Wo = (const float*)d_in[9];
  const float* bo = (const float*)d_in[10];

  const size_t per = (size_t)BATCH * HEADS * SEQ * HEAD_DIM;  // 8.4M elems
  bf16_t* qw = (bf16_t*)d_ws;
  bf16_t* kw = qw + per;
  bf16_t* vw = kw + per;
  bf16_t* aw = vw + per;

  dim3 gproj(MTOT / 64, HIDDEN / 64);   // 128 x 16
  dim3 blk(128);
  mha_proj_qkv<<<gproj, blk, 0, stream>>>(queries, Wq, bq, qw);
  mha_proj_qkv<<<gproj, blk, 0, stream>>>(keys,    Wk, bk, kw);
  mha_proj_qkv<<<gproj, blk, 0, stream>>>(values,  Wv, bv, vw);

  dim3 gattn(SEQ / 64, BATCH * HEADS);  // 32 x 64
  mha_attn<<<gattn, blk, 0, stream>>>(qw, kw, vw, aw);

  mha_proj_out<<<gproj, blk, 0, stream>>>(aw, Wo, bo, (float*)d_out);
}
